// NeighborhoodCrossAttention_82231443849835
// MI455X (gfx1250) — matrix operand order repaired and checked by reading, not compiled, not run on hardware
//
#include <hip/hip_runtime.h>
#include <hip/hip_bf16.h>

// ---------- types for CDNA5 WMMA ----------
typedef __bf16 v16bf __attribute__((ext_vector_type(16)));
typedef float  v8f   __attribute__((ext_vector_type(8)));

#define DIMC   128
#define HEADS  8
#define HDIM   16
#define HSZ    32
#define WSZ    32
#define ZSZ    8
#define NTOK   8192     // 32*32*8
#define KCUBE  125      // 5^3

// ---------- workspace layout (bytes) ----------
#define OFF_STATS 0                                   // 4*128 floats = 2048 B
#define OFF_AQ    2048                                // ushort [8192][128] = 2 MB
#define OFF_AS    (OFF_AQ + NTOK*DIMC*2)              // ushort [8192][128] = 2 MB
#define OFF_WT    (OFF_AS + NTOK*DIMC*2)              // 4 * ushort[128][128] = 128 KB
#define OFF_Q     (OFF_WT + 4*DIMC*DIMC*2)            // float [8192][128] = 4 MB
#define OFF_K     (OFF_Q + NTOK*DIMC*4)
#define OFF_V     (OFF_K + NTOK*DIMC*4)
#define OFF_AO    (OFF_V + NTOK*DIMC*4)               // ushort [8192][128] = 2 MB

__device__ __forceinline__ unsigned short f2bf_bits(float f) {
  unsigned u = __float_as_uint(f);
  u += 0x7FFFu + ((u >> 16) & 1u);      // round-to-nearest-even
  return (unsigned short)(u >> 16);
}

// ---------------- kernel 1: instance-norm statistics ----------------
__global__ void nca_stats_kernel(const float* __restrict__ x,
                                 const float* __restrict__ skip,
                                 float* __restrict__ stats) {
  __shared__ float s1[256], s2[256];
  int b = blockIdx.x;              // 0..127 -> x channels, 128..255 -> skip channels
  int t = threadIdx.x;
  bool isx = b < 128;
  int c = isx ? b : b - 128;
  const float* src = isx ? (x + c * 1024) : (skip + c * 8192);
  int cnt = isx ? 1024 : 8192;
  float s = 0.f, q = 0.f;
  for (int i = t; i < cnt; i += 256) { float v = src[i]; s += v; q += v * v; }
  s1[t] = s; s2[t] = q;
  __syncthreads();
  for (int o = 128; o > 0; o >>= 1) {
    if (t < o) { s1[t] += s1[t + o]; s2[t] += s2[t + o]; }
    __syncthreads();
  }
  if (t == 0) {
    float mu  = s1[0] / (float)cnt;
    float var = s2[0] / (float)cnt - mu * mu;
    float* dst = stats + (isx ? 0 : 256);
    dst[c]       = mu;
    dst[128 + c] = rsqrtf(var + 1e-5f);
  }
}

// ------- kernel 2: fused norm + upsample + tokenize -> bf16 A matrices -------
__global__ void nca_pack_kernel(const float* __restrict__ x,
                                const float* __restrict__ skip,
                                const float* __restrict__ stats,
                                unsigned short* __restrict__ aq,
                                unsigned short* __restrict__ as_) {
  int idx = blockIdx.x * 256 + threadIdx.x;   // NTOK*DIMC = 1,048,576
  int n = idx >> 7, c = idx & 127;
  int h = n >> 8, w = (n >> 3) & 31, z = n & 7;
  float mux = stats[c],       rx = stats[128 + c];
  float mus = stats[256 + c], rs = stats[384 + c];
  float xv = x[c * 1024 + (h >> 1) * 64 + (w >> 1) * 4 + (z >> 1)];  // 2x upsample
  float sv = skip[c * 8192 + n];
  aq[n * 128 + c]  = f2bf_bits((xv - mux) * rx);
  as_[n * 128 + c] = f2bf_bits((sv - mus) * rs);
}

// ------- kernel 3: transpose + bf16-convert the 4 weight matrices -------
__global__ void nca_wconv_kernel(const float* __restrict__ Wq,
                                 const float* __restrict__ Wk,
                                 const float* __restrict__ Wv,
                                 const float* __restrict__ Wo,
                                 unsigned short* __restrict__ wt) {
  int idx = blockIdx.x * 256 + threadIdx.x;   // 4*128*128 = 65536
  int m = idx >> 14;
  int k = (idx >> 7) & 127;
  int n = idx & 127;
  const float* W = (m == 0) ? Wq : (m == 1) ? Wk : (m == 2) ? Wv : Wo;
  wt[m * 16384 + n * 128 + k] = f2bf_bits(W[k * 128 + n]);   // WT[n][k] = W[k][n]
}

// ---------- WMMA fragment loaders (ISA 7.12.2 bf16 layouts) ----------
// A 16x32 bf16: lane(r=l&15) holds row M=r; elems 0..7 -> K = hi*8+e,
//               elems 8..15 -> K = 16 + hi*8 + (e-8), hi = l>>4.
__device__ __forceinline__ v16bf nca_load_frag_A(const unsigned short* __restrict__ A,
                                                 int row0, int k0, int lane) {
  int r = lane & 15, hi = lane >> 4;
  const unsigned short* p = A + (size_t)(row0 + r) * 128 + k0 + hi * 8;
  union { uint4 u[2]; v16bf v; } t;
  t.u[0] = *(const uint4*)(p);
  t.u[1] = *(const uint4*)(p + 16);
  return t.v;
}
// B 32x16 bf16 (from transposed weight WT[n][k]): lane N = l&15,
// elems 0..7 -> K = (l>>4)*8 + e, elems 8..15 -> K = 16 + (l>>4)*8 + (e-8).
__device__ __forceinline__ v16bf nca_load_frag_B(const unsigned short* __restrict__ WT,
                                                 int col0, int k0, int lane) {
  int nn = lane & 15, hi = lane >> 4;
  const unsigned short* p = WT + (size_t)(col0 + nn) * 128 + k0 + hi * 8;
  union { uint4 u[2]; v16bf v; } t;
  t.u[0] = *(const uint4*)(p);
  t.u[1] = *(const uint4*)(p + 16);
  return t.v;
}

// ---------------- kernel 4: fused Q/K/V projection GEMMs ----------------
// grid (512, 3): y selects Q / K / V. One wave per 16x16 output tile.
__global__ void nca_gemm_qkv_kernel(const unsigned short* __restrict__ aq,
                                    const unsigned short* __restrict__ as_,
                                    const unsigned short* __restrict__ wt,
                                    const float* __restrict__ bq,
                                    const float* __restrict__ bk,
                                    const float* __restrict__ bv,
                                    float* __restrict__ Qo,
                                    float* __restrict__ Ko,
                                    float* __restrict__ Vo) {
  int which = blockIdx.y;
  const unsigned short* A = (which == 0) ? aq : as_;
  const unsigned short* B = wt + which * 16384;
  const float* bias = (which == 0) ? bq : (which == 1) ? bk : bv;
  float* out = (which == 0) ? Qo : (which == 1) ? Ko : Vo;
  float scale = (which == 0) ? 0.25f : 1.0f;   // HEAD_DIM^-0.5 applied to Q

  int lane = threadIdx.x & 31;
  int wv   = threadIdx.x >> 5;
  int tile = blockIdx.x * 8 + wv;              // 4096 tiles = 512 x 8
  int tm = (tile >> 3) * 16;
  int tn = (tile & 7) * 16;

  v8f acc = {};
#pragma unroll
  for (int kk = 0; kk < 128; kk += 32) {
    v16bf a = nca_load_frag_A(A, tm, kk, lane);
    v16bf b = nca_load_frag_B(B, tn, kk, lane);
    acc = __builtin_amdgcn_wmma_f32_16x16x32_bf16(false, a, false, b,
                                                  (short)0, acc, false, false);
  }
  // C/D: elem i -> M = 8*(lane>>4)+i, N = lane&15
  int col   = tn + (lane & 15);
  int rbase = tm + (lane >> 4) * 8;
  float bc = bias[col];
#pragma unroll
  for (int i = 0; i < 8; ++i)
    out[(size_t)(rbase + i) * 128 + col] = (acc[i] + bc) * scale;
}

// ---------------- kernel 5: neighborhood attention ----------------
// One wave per (token, head). Lanes = 2 groups x 16 dims.
__global__ void nca_attn_kernel(const float* __restrict__ Q,
                                const float* __restrict__ Km,
                                const float* __restrict__ Vm,
                                const float* __restrict__ rpb,
                                unsigned short* __restrict__ ao) {
  __shared__ float sl[8][128];
  int lane = threadIdx.x & 31;
  int wv   = threadIdx.x >> 5;
  int unit = blockIdx.x * 8 + wv;   // 65536 units
  int n = unit >> 3;
  int head = unit & 7;
  int h = n >> 8, w = (n >> 3) & 31, z = n & 7;
  int sh = min(max(h - 2, 0), 27);
  int sw = min(max(w - 2, 0), 27);
  int sz = min(max(z - 2, 0), 3);
  int g = lane >> 4, d = lane & 15;
  int chan = head * 16 + d;
  float qd = Q[(size_t)n * 128 + chan];
  const float* rp = rpb + head * 729;          // (9,9,9)
  int bh = sh - h + 4, bw = sw - w + 4, bz = sz - z + 4;

  // pass 1: logits = q . k_nbr + bias
  for (int j = 0; j < KCUBE; j += 2) {
    int jj = j + g;
    bool valid = jj < KCUBE;
    int a = 0, b = 0, c = 0;
    float p = 0.f;
    if (valid) {
      a = jj / 25; int rem = jj - a * 25; b = rem / 5; c = rem - b * 5;
      int nbr = (sh + a) * 256 + (sw + b) * 8 + (sz + c);
      p = qd * Km[(size_t)nbr * 128 + chan];
    }
    p += __shfl_xor(p, 1, 32);
    p += __shfl_xor(p, 2, 32);
    p += __shfl_xor(p, 4, 32);
    p += __shfl_xor(p, 8, 32);
    if (valid && d == 0)
      sl[wv][jj] = p + rp[(bh + a) * 81 + (bw + b) * 9 + (bz + c)];
  }
  __syncthreads();

  // pass 2: softmax (wave-wide max then exp-sum)
  float m = -3.0e38f;
  for (int j = lane; j < KCUBE; j += 32) m = fmaxf(m, sl[wv][j]);
  for (int o = 16; o > 0; o >>= 1) m = fmaxf(m, __shfl_xor(m, o, 32));
  float s = 0.f;
  for (int j = lane; j < KCUBE; j += 32) {
    float e = __expf(sl[wv][j] - m);
    sl[wv][j] = e;
    s += e;
  }
  for (int o = 16; o > 0; o >>= 1) s += __shfl_xor(s, o, 32);
  float inv = 1.0f / s;
  __syncthreads();

  // pass 3: out_d = sum_j w_j * v[j][d]
  float acc = 0.f;
  for (int j = 0; j < KCUBE; j += 2) {
    int jj = j + g;
    if (jj < KCUBE) {
      int a = jj / 25; int rem = jj - a * 25; int b = rem / 5; int c = rem - b * 5;
      int nbr = (sh + a) * 256 + (sw + b) * 8 + (sz + c);
      acc += sl[wv][jj] * Vm[(size_t)nbr * 128 + chan];
    }
  }
  acc += __shfl_xor(acc, 16, 32);
  if (g == 0) ao[(size_t)n * 128 + chan] = f2bf_bits(acc * inv);
}

// ---------------- kernel 6: output projection GEMM + final transpose ----------------
__global__ void nca_gemm_out_kernel(const unsigned short* __restrict__ ao,
                                    const unsigned short* __restrict__ wt,
                                    const float* __restrict__ bo,
                                    float* __restrict__ outp) {
  int lane = threadIdx.x & 31;
  int wv   = threadIdx.x >> 5;
  int tile = blockIdx.x * 8 + wv;
  int tm = (tile >> 3) * 16;
  int tn = (tile & 7) * 16;
  const unsigned short* B = wt + 3 * 16384;    // WoT

  v8f acc = {};
#pragma unroll
  for (int kk = 0; kk < 128; kk += 32) {
    v16bf a = nca_load_frag_A(ao, tm, kk, lane);
    v16bf b = nca_load_frag_B(B, tn, kk, lane);
    acc = __builtin_amdgcn_wmma_f32_16x16x32_bf16(false, a, false, b,
                                                  (short)0, acc, false, false);
  }
  int col   = tn + (lane & 15);
  int rbase = tm + (lane >> 4) * 8;
  float bc = bo[col];
  // output layout (1,128,32,32,8): d_out[c*8192 + n]; consecutive i = consecutive n
#pragma unroll
  for (int i = 0; i < 8; ++i)
    outp[(size_t)col * 8192 + rbase + i] = acc[i] + bc;
}

extern "C" void kernel_launch(void* const* d_in, const int* in_sizes, int n_in,
                              void* d_out, int out_size, void* d_ws, size_t ws_size,
                              hipStream_t stream) {
  const float* x    = (const float*)d_in[0];
  const float* skip = (const float*)d_in[1];
  const float* Wq   = (const float*)d_in[2];
  const float* bq   = (const float*)d_in[3];
  const float* Wk   = (const float*)d_in[4];
  const float* bk   = (const float*)d_in[5];
  const float* Wv   = (const float*)d_in[6];
  const float* bv   = (const float*)d_in[7];
  const float* rpb  = (const float*)d_in[8];
  const float* Wo   = (const float*)d_in[9];
  const float* bo   = (const float*)d_in[10];

  char* ws = (char*)d_ws;
  float*          stats = (float*)(ws + OFF_STATS);
  unsigned short* aq    = (unsigned short*)(ws + OFF_AQ);
  unsigned short* as_   = (unsigned short*)(ws + OFF_AS);
  unsigned short* wt    = (unsigned short*)(ws + OFF_WT);
  float*          Qp    = (float*)(ws + OFF_Q);
  float*          Kp    = (float*)(ws + OFF_K);
  float*          Vp    = (float*)(ws + OFF_V);
  unsigned short* ao    = (unsigned short*)(ws + OFF_AO);

  nca_stats_kernel<<<256, 256, 0, stream>>>(x, skip, stats);
  nca_pack_kernel<<<4096, 256, 0, stream>>>(x, skip, stats, aq, as_);
  nca_wconv_kernel<<<256, 256, 0, stream>>>(Wq, Wk, Wv, Wo, wt);
  nca_gemm_qkv_kernel<<<dim3(512, 3), 256, 0, stream>>>(aq, as_, wt, bq, bk, bv,
                                                        Qp, Kp, Vp);
  nca_attn_kernel<<<8192, 256, 0, stream>>>(Qp, Kp, Vp, rpb, ao);
  nca_gemm_out_kernel<<<512, 256, 0, stream>>>(ao, wt, bo, (float*)d_out);
}
